// PyGGNNEncoder_6313601925545
// MI455X (gfx1250) — compile-verified, matrix operand order checked by reading, not run on hardware
//
#include <hip/hip_runtime.h>
#include <hip/hip_bf16.h>
#include <math.h>

#define HDIM   128
#define NHEADS 4
#define DHEAD  32
#define NLAYERS 3
#define NB     64
#define LDSH   136   // f16 row stride in LDS (multiple of 8 halves -> 16B aligned vectors)

typedef __attribute__((ext_vector_type(16))) _Float16 v16h;
typedef __attribute__((ext_vector_type(8)))  _Float16 v8h;
typedef __attribute__((ext_vector_type(8)))  float    v8f;
typedef int async_i4 __attribute__((vector_size(16)));

#if defined(__has_builtin)
#if __has_builtin(__builtin_amdgcn_global_load_async_to_lds_b128)
#define USE_ASYNC_LDS 1
#endif
#endif

__device__ __forceinline__ float silu_f(float v) { return v / (1.f + __expf(-v)); }

__device__ __forceinline__ void atomicMaxF(float* addr, float val) {
    if (val >= 0.f) atomicMax((int*)addr, __float_as_int(val));
    else            atomicMin((unsigned int*)addr, __float_as_uint(val));
}

// Pre-convert a 128x128 f32 weight (row-major [k][n]) into transposed f16 [n][k].
__global__ __launch_bounds__(256) void wcvt_kernel(const float* __restrict__ W,
                                                   _Float16* __restrict__ Wt)
{
    int i = blockIdx.x * blockDim.x + threadIdx.x;
    if (i >= HDIM * HDIM) return;
    int n = i >> 7, k = i & 127;
    Wt[n * HDIM + k] = (_Float16)W[k * HDIM + n];
}

// ---------------------------------------------------------------------------
// Fused GEMM: out[r,:] = A[r,:128] @ W[128,:128]  (+ bias, LayerNorm, SiLU)
// One block = 16 output rows. 8 waves, each wave owns a 16x16 WMMA tile.
// A is staged f32->f16; the pre-transposed f16 weight is staged with
// async-to-LDS copies when the toolchain exposes them.
// ---------------------------------------------------------------------------
__global__ __launch_bounds__(256) void wmma_gemm_kernel(
    const float* __restrict__ A, const _Float16* __restrict__ Wt,
    const float* __restrict__ bias, const float* __restrict__ gamma,
    const float* __restrict__ beta, float* __restrict__ out,
    int n_rows, int fuse_ln)
{
    __shared__ __align__(16) _Float16 a_sh[16 * LDSH];
    __shared__ __align__(16) _Float16 w_sh[HDIM * LDSH];  // [n][k]
    __shared__ float o_sh[16 * HDIM];
    __shared__ float s_mu[16], s_rstd[16];

    const int tid  = threadIdx.x;
    const int wave = tid >> 5;
    const int lane = tid & 31;
    const int hs   = lane >> 4;    // half-select (K interleave)
    const int l16  = lane & 15;
    const int row0 = blockIdx.x * 16;

    // Stage W (f16, pre-transposed): 128 rows x 128 halves, 16B chunks
#ifdef USE_ASYNC_LDS
    for (int i = tid; i < HDIM * 16; i += 256) {
        int n = i >> 4, ck = (i & 15) * 8;
        __builtin_amdgcn_global_load_async_to_lds_b128(
            (__attribute__((address_space(1))) async_i4*)(Wt + n * HDIM + ck),
            (__attribute__((address_space(3))) async_i4*)(&w_sh[n * LDSH + ck]),
            0, 0);
    }
#else
    for (int i = tid; i < HDIM * 16; i += 256) {
        int n = i >> 4, ck = (i & 15) * 8;
        *(v8h*)&w_sh[n * LDSH + ck] = *(const v8h*)(Wt + n * HDIM + ck);
    }
#endif

    // Stage A strip (16 x 128), f32 -> f16
    for (int i = tid; i < 16 * HDIM; i += 256) {
        int r = i >> 7, k = i & 127;
        float v = (row0 + r < n_rows) ? A[(size_t)(row0 + r) * HDIM + k] : 0.0f;
        a_sh[r * LDSH + k] = (_Float16)v;
    }

#ifdef USE_ASYNC_LDS
#if __has_builtin(__builtin_amdgcn_s_wait_asynccnt)
    __builtin_amdgcn_s_wait_asynccnt(0);
#else
    asm volatile("s_wait_asynccnt 0" ::: "memory");
#endif
#endif
    __syncthreads();

    v8f c = {};
    const int ncol = wave * 16 + l16;
    #pragma unroll
    for (int kk = 0; kk < HDIM; kk += 32) {
        // 16-bit operand layout: lanes 0-15 hold K kk..kk+7 & kk+16..kk+23,
        // lanes 16-31 hold K kk+8..kk+15 & kk+24..kk+31.
        const _Float16* ap = &a_sh[l16  * LDSH + kk + hs * 8];
        const _Float16* bp = &w_sh[ncol * LDSH + kk + hs * 8];
        v8h alo = *(const v8h*)(ap);
        v8h ahi = *(const v8h*)(ap + 16);
        v8h blo = *(const v8h*)(bp);
        v8h bhi = *(const v8h*)(bp + 16);
        v16h a, b;
        #pragma unroll
        for (int i = 0; i < 8; ++i) {
            a[i] = alo[i]; a[8 + i] = ahi[i];
            b[i] = blo[i]; b[8 + i] = bhi[i];
        }
        c = __builtin_amdgcn_wmma_f32_16x16x32_f16(false, a, false, b,
                                                   (short)0, c, false, false);
    }

    if (!fuse_ln) {
        // C/D layout: VGPR r holds M=r (lanes 0-15), M=r+8 (lanes 16-31)
        #pragma unroll
        for (int r = 0; r < 8; ++r) {
            int m = r + hs * 8;
            if (row0 + m < n_rows)
                out[(size_t)(row0 + m) * HDIM + ncol] = c[r];
        }
    } else {
        #pragma unroll
        for (int r = 0; r < 8; ++r) {
            int m = r + hs * 8;
            o_sh[m * HDIM + ncol] = c[r] + bias[ncol];
        }
        __syncthreads();
        if (tid < 16) {
            const float* rowp = &o_sh[tid * HDIM];
            float mu = 0.f;
            for (int k = 0; k < HDIM; ++k) mu += rowp[k];
            mu *= (1.f / HDIM);
            float var = 0.f;
            for (int k = 0; k < HDIM; ++k) { float d = rowp[k] - mu; var += d * d; }
            var *= (1.f / HDIM);
            s_mu[tid]   = mu;
            s_rstd[tid] = rsqrtf(var + 1e-5f);
        }
        __syncthreads();
        for (int i = tid; i < 16 * HDIM; i += 256) {
            int r = i >> 7, k = i & 127;
            if (row0 + r < n_rows) {
                float y = (o_sh[i] - s_mu[r]) * s_rstd[r] * gamma[k] + beta[k];
                out[(size_t)(row0 + r) * HDIM + k] = silu_f(y);
            }
        }
    }
}

// One wave32 per node: coalesced row read, 4 heads reduced with shuffles.
__global__ __launch_bounds__(256) void attn_proj_kernel(
    const float* __restrict__ hh, const float* __restrict__ att_s,
    const float* __restrict__ att_d, float* __restrict__ a_s,
    float* __restrict__ a_d, int n_nodes)
{
    int wave = threadIdx.x >> 5, lane = threadIdx.x & 31;
    int n = blockIdx.x * 8 + wave;
    if (n >= n_nodes) return;
    const float* hp = hh + (size_t)n * HDIM;
    float ps[NHEADS], pd[NHEADS];
    #pragma unroll
    for (int j = 0; j < NHEADS; ++j) {
        int k = j * DHEAD + lane;
        float v = hp[k];
        ps[j] = v * att_s[k];
        pd[j] = v * att_d[k];
    }
    #pragma unroll
    for (int j = 0; j < NHEADS; ++j) {
        #pragma unroll
        for (int m = 16; m > 0; m >>= 1) {
            ps[j] += __shfl_xor(ps[j], m, 32);
            pd[j] += __shfl_xor(pd[j], m, 32);
        }
    }
    if (lane == 0) {
        #pragma unroll
        for (int j = 0; j < NHEADS; ++j) {
            a_s[n * NHEADS + j] = ps[j];
            a_d[n * NHEADS + j] = pd[j];
        }
    }
}

__global__ __launch_bounds__(256) void fill_kernel(float* __restrict__ p, float v, long count)
{
    long i = (long)blockIdx.x * blockDim.x + threadIdx.x;
    if (i < count) p[i] = v;
}

// Pass 1: e = leaky_relu(a_s[src]+a_d[dst]); store logits; atomic segment-max
__global__ __launch_bounds__(256) void edge_logit_kernel(
    const int* __restrict__ ei, const float* __restrict__ a_s,
    const float* __restrict__ a_d, float* __restrict__ ebuf,
    float* __restrict__ mbuf, int E0, int ET)
{
    int e = blockIdx.x * blockDim.x + threadIdx.x;
    if (e >= ET) return;
    int s, d;
    if (e < E0) { s = ei[e]; d = ei[E0 + e]; } else { s = e - E0; d = s; }
    #pragma unroll
    for (int h = 0; h < NHEADS; ++h) {
        float ev = a_s[s * NHEADS + h] + a_d[d * NHEADS + h];
        ev = (ev > 0.f) ? ev : 0.2f * ev;
        ebuf[(size_t)e * NHEADS + h] = ev;
        atomicMaxF(&mbuf[d * NHEADS + h], ev);
    }
}

// Pass 2: alpha = exp(e - max[dst]); store; atomic segment-sum of denominators
__global__ __launch_bounds__(256) void edge_exp_kernel(
    const int* __restrict__ ei, float* __restrict__ ebuf,
    const float* __restrict__ mbuf, float* __restrict__ denom, int E0, int ET)
{
    int e = blockIdx.x * blockDim.x + threadIdx.x;
    if (e >= ET) return;
    int d = (e < E0) ? ei[E0 + e] : (e - E0);
    #pragma unroll
    for (int h = 0; h < NHEADS; ++h) {
        float al = __expf(ebuf[(size_t)e * NHEADS + h] - mbuf[d * NHEADS + h]);
        ebuf[(size_t)e * NHEADS + h] = al;
        atomicAdd(&denom[d * NHEADS + h], al);
    }
}

// Pass 3: one wave32 per edge; lane k handles dim head*32+k -> coalesced
// gather of hh[src] and coalesced atomic adds into accum[dst].
__global__ __launch_bounds__(256) void edge_scatter_kernel(
    const int* __restrict__ ei, const float* __restrict__ ebuf,
    const float* __restrict__ denom, const float* __restrict__ hh,
    float* __restrict__ accum, int E0, int ET)
{
    int wave = threadIdx.x >> 5, lane = threadIdx.x & 31;
    int e = blockIdx.x * 8 + wave;
    if (e >= ET) return;
    int s, d;
    if (e < E0) { s = ei[e]; d = ei[E0 + e]; } else { s = e - E0; d = s; }
    const float* hp = hh    + (size_t)s * HDIM;
    float*       op = accum + (size_t)d * HDIM;
    #pragma unroll
    for (int j = 0; j < NHEADS; ++j) {
        float w = ebuf[(size_t)e * NHEADS + j] / denom[d * NHEADS + j];
        int k = j * DHEAD + lane;
        atomicAdd(&op[k], hp[k] * w);
    }
}

// h = silu(LN(accum + bg)) + h   — one wave32 per node, shuffle reductions
__global__ __launch_bounds__(256) void node_update_kernel(
    const float* __restrict__ accum, const float* __restrict__ bias,
    const float* __restrict__ gamma, const float* __restrict__ beta,
    float* __restrict__ h, int n_nodes)
{
    int wave = threadIdx.x >> 5, lane = threadIdx.x & 31;
    int n = blockIdx.x * 8 + wave;
    if (n >= n_nodes) return;
    const float* ap = accum + (size_t)n * HDIM;
    float v[4]; float s = 0.f;
    #pragma unroll
    for (int j = 0; j < 4; ++j) {
        v[j] = ap[lane + 32 * j] + bias[lane + 32 * j];
        s += v[j];
    }
    #pragma unroll
    for (int m = 16; m > 0; m >>= 1) s += __shfl_xor(s, m, 32);
    float mu = s * (1.f / HDIM);
    float q = 0.f;
    #pragma unroll
    for (int j = 0; j < 4; ++j) { float dd = v[j] - mu; q += dd * dd; }
    #pragma unroll
    for (int m = 16; m > 0; m >>= 1) q += __shfl_xor(q, m, 32);
    float rstd = rsqrtf(q * (1.f / HDIM) + 1e-5f);
    float* hp = h + (size_t)n * HDIM;
    #pragma unroll
    for (int j = 0; j < 4; ++j) {
        int k = lane + 32 * j;
        float y = (v[j] - mu) * rstd * gamma[k] + beta[k];
        hp[k] = silu_f(y) + hp[k];
    }
}

__global__ __launch_bounds__(256) void pool_accum_kernel(
    const float* __restrict__ h, const int* __restrict__ batch,
    float* __restrict__ pooled, float* __restrict__ cnt, int n_nodes)
{
    long i = (long)blockIdx.x * blockDim.x + threadIdx.x;
    if (i >= (long)n_nodes * HDIM) return;
    int n = (int)(i >> 7), k = (int)(i & 127);
    int b = batch[n];
    atomicAdd(&pooled[b * HDIM + k], h[i]);
    if (k == 0) atomicAdd(&cnt[b], 1.0f);
}

__global__ __launch_bounds__(256) void pool_div_kernel(
    float* __restrict__ pooled, const float* __restrict__ cnt)
{
    int i = blockIdx.x * blockDim.x + threadIdx.x;
    if (i >= NB * HDIM) return;
    pooled[i] /= fmaxf(cnt[i >> 7], 1.0f);
}

extern "C" void kernel_launch(void* const* d_in, const int* in_sizes, int n_in,
                              void* d_out, int out_size, void* d_ws, size_t ws_size,
                              hipStream_t stream)
{
    const float* x      = (const float*)d_in[0];
    const int*   ei     = (const int*)  d_in[1];
    const int*   batch  = (const int*)  d_in[2];
    const float* W0     = (const float*)d_in[3];
    const float* b0     = (const float*)d_in[4];
    const float* g0     = (const float*)d_in[5];
    const float* be0    = (const float*)d_in[6];
    const float* Wg     = (const float*)d_in[7];
    const float* att_s  = (const float*)d_in[8];
    const float* att_d  = (const float*)d_in[9];
    const float* bg     = (const float*)d_in[10];
    const float* gg     = (const float*)d_in[11];
    const float* bgg    = (const float*)d_in[12];
    const float* Wp     = (const float*)d_in[13];
    const float* bp     = (const float*)d_in[14];
    const float* gp     = (const float*)d_in[15];
    const float* bep    = (const float*)d_in[16];
    float* out = (float*)d_out;

    const int N  = in_sizes[0] / HDIM;
    const int E0 = in_sizes[1] / 2;
    const int ET = E0 + N;     // + self loops

    // workspace carve-out (256B aligned)
    char* wsp = (char*)d_ws;
    auto carve = [&](size_t bytes) -> char* {
        char* p = wsp; wsp += (bytes + 255) & ~(size_t)255; return p;
    };
    float*     h      = (float*)carve((size_t)N * HDIM * 4);
    float*     hh     = (float*)carve((size_t)N * HDIM * 4);
    float*     accum  = (float*)carve((size_t)N * HDIM * 4);
    float*     a_src  = (float*)carve((size_t)N * NHEADS * 4);
    float*     a_dst  = (float*)carve((size_t)N * NHEADS * 4);
    float*     mbuf   = (float*)carve((size_t)N * NHEADS * 4);
    float*     denom  = (float*)carve((size_t)N * NHEADS * 4);
    float*     ebuf   = (float*)carve((size_t)ET * NHEADS * 4);
    float*     pooled = (float*)carve((size_t)NB * HDIM * 4);
    float*     cnt    = (float*)carve((size_t)NB * 4);
    _Float16*  wt     = (_Float16*)carve((size_t)5 * HDIM * HDIM * 2);

    _Float16* wt0 = wt;                       // encoder
    _Float16* wtg = wt + (size_t)HDIM * HDIM; // 3 GAT layers
    _Float16* wtp = wt + (size_t)4 * HDIM * HDIM;

    dim3 blk(256);
    const int wcvt_blocks = (HDIM * HDIM + 255) / 256;
    const int gemm_blocks = (N + 15) / 16;

    // Pre-convert all weights: f32 [k][n] -> f16 transposed [n][k]
    wcvt_kernel<<<wcvt_blocks, blk, 0, stream>>>(W0, wt0);
    for (int l = 0; l < NLAYERS; ++l)
        wcvt_kernel<<<wcvt_blocks, blk, 0, stream>>>(
            Wg + (size_t)l * HDIM * HDIM, wtg + (size_t)l * HDIM * HDIM);
    wcvt_kernel<<<wcvt_blocks, blk, 0, stream>>>(Wp, wtp);

    // Node encoder: h = silu(LN(x @ W0 + b0))
    wmma_gemm_kernel<<<gemm_blocks, blk, 0, stream>>>(x, wt0, b0, g0, be0, h, N, 1);

    for (int l = 0; l < NLAYERS; ++l) {
        wmma_gemm_kernel<<<gemm_blocks, blk, 0, stream>>>(
            h, wtg + (size_t)l * HDIM * HDIM, nullptr, nullptr, nullptr, hh, N, 0);
        attn_proj_kernel<<<(N + 7) / 8, blk, 0, stream>>>(
            hh, att_s + (size_t)l * NHEADS * DHEAD, att_d + (size_t)l * NHEADS * DHEAD,
            a_src, a_dst, N);
        fill_kernel<<<(int)(((long)N * NHEADS + 255) / 256), blk, 0, stream>>>(mbuf, -1e30f, (long)N * NHEADS);
        fill_kernel<<<(int)(((long)N * NHEADS + 255) / 256), blk, 0, stream>>>(denom, 0.f, (long)N * NHEADS);
        fill_kernel<<<(int)(((long)N * HDIM + 255) / 256), blk, 0, stream>>>(accum, 0.f, (long)N * HDIM);
        edge_logit_kernel<<<(ET + 255) / 256, blk, 0, stream>>>(ei, a_src, a_dst, ebuf, mbuf, E0, ET);
        edge_exp_kernel<<<(ET + 255) / 256, blk, 0, stream>>>(ei, ebuf, mbuf, denom, E0, ET);
        edge_scatter_kernel<<<(ET + 7) / 8, blk, 0, stream>>>(ei, ebuf, denom, hh, accum, E0, ET);
        node_update_kernel<<<(N + 7) / 8, blk, 0, stream>>>(
            accum, bg + (size_t)l * HDIM, gg + (size_t)l * HDIM, bgg + (size_t)l * HDIM, h, N);
    }

    // Global mean pool per graph, then final projection + LN + SiLU
    fill_kernel<<<(NB * HDIM + 255) / 256, blk, 0, stream>>>(pooled, 0.f, NB * HDIM);
    fill_kernel<<<1, blk, 0, stream>>>(cnt, 0.f, NB);
    pool_accum_kernel<<<(int)(((long)N * HDIM + 255) / 256), blk, 0, stream>>>(h, batch, pooled, cnt, N);
    pool_div_kernel<<<(NB * HDIM + 255) / 256, blk, 0, stream>>>(pooled, cnt);
    wmma_gemm_kernel<<<(NB + 15) / 16, blk, 0, stream>>>(pooled, wtp, bp, gp, bep, out, NB, 1);
}
